// CrfRnnLayerAll_66168266162189
// MI455X (gfx1250) — matrix operand order repaired and checked by reading, not compiled, not run on hardware
//
#include <hip/hip_runtime.h>
#include <hip/hip_bf16.h>
#include <math.h>
#include <stdint.h>

// ---------------------------------------------------------------------------
// CRF-RNN mean-field loop for MI455X (gfx1250, wave32).
// Memory-bound: minimize passes, fuse GEMM with the q-update, use f16 WMMA
// 16x16x32 (21 channels padded to K=32) with fp32 accumulation, and use the
// CDNA5 async global->LDS path for the blur tile fills.
// ---------------------------------------------------------------------------

#define CCH   21
#define CPAD  32
#define HH    512
#define WW    512
#define NPIX  (HH * WW)          // 262144
#define CN    (CCH * NPIX)       // 5505024
#define RAD   7
#define TAPS  15
#define NUM_ITER 5

#define SIGMA_SP 3.0f            // THETA_GAMMA
#define SIGMA_BI 10.0f           // THETA_ALPHA / 16

typedef _Float16 v16h __attribute__((ext_vector_type(16)));
typedef float    v8f  __attribute__((ext_vector_type(8)));

__device__ __forceinline__ void gauss_weights(float sigma, float* w) {
  float s = 0.f;
#pragma unroll
  for (int i = 0; i < TAPS; ++i) {
    float x = (float)(i - RAD);
    w[i] = __expf(-0.5f * x * x / (sigma * sigma));
    s += w[i];
  }
  float inv = 1.f / s;
#pragma unroll
  for (int i = 0; i < TAPS; ++i) w[i] *= inv;
}

// generic pointer to __shared__ has the LDS byte offset in its low 32 bits
__device__ __forceinline__ unsigned ldsAddr(const void* p) {
  return (unsigned)(uintptr_t)p;
}

__device__ __forceinline__ void asyncLoadLdsB32(unsigned lds, const float* g) {
  asm volatile("global_load_async_to_lds_b32 %0, %1, off"
               :: "v"(lds), "v"((unsigned long long)(uintptr_t)g)
               : "memory");
}

__device__ __forceinline__ void waitAsync0() {
  asm volatile("s_wait_asynccnt 0x0" ::: "memory");
}

// --------------------------- preparation -----------------------------------

// guide(h,w) = exp(-0.5 * sum((rgb/255)^2) / theta_beta^2)
__global__ void kPrepGuide(const float* __restrict__ rgb, float* __restrict__ guide) {
  int p = blockIdx.x * blockDim.x + threadIdx.x;
  float r = rgb[(size_t)p * 3 + 0] * (1.0f / 255.0f);
  float g = rgb[(size_t)p * 3 + 1] * (1.0f / 255.0f);
  float b = rgb[(size_t)p * 3 + 2] * (1.0f / 255.0f);
  guide[p] = __expf(-0.5f * (r * r + g * g + b * b) / 9.0f);
}

// HWC -> planar CHW copy of unaries into u, and q = u
__global__ void kPrepU(const float* __restrict__ unaries, float* __restrict__ u,
                       float* __restrict__ q) {
  size_t idx = (size_t)blockIdx.x * blockDim.x + threadIdx.x;
  int c = (int)(idx / NPIX);
  int p = (int)(idx % NPIX);
  float v = unaries[(size_t)p * CCH + c];
  u[idx] = v;
  q[idx] = v;
}

// M1 = compat @ Ws, M2 = compat @ Wb, zero-padded to 32x32, stored f16 row-major
__global__ void kMatPrep(const float* __restrict__ compat,
                         const float* __restrict__ Ws,
                         const float* __restrict__ Wb,
                         _Float16* __restrict__ M1, _Float16* __restrict__ M2) {
  int tid = blockIdx.x * blockDim.x + threadIdx.x;   // 0 .. 2047
  int which = tid >> 10;
  int rem = tid & 1023;
  int r = rem >> 5, c = rem & 31;
  const float* W = which ? Wb : Ws;
  float acc = 0.f;
  if (r < CCH && c < CCH) {
#pragma unroll
    for (int k = 0; k < CCH; ++k) acc += compat[r * CCH + k] * W[k * CCH + c];
  }
  if (which) M2[rem] = (_Float16)acc; else M1[rem] = (_Float16)acc;
}

// blur(ones) with zero padding is separable: product of partial tap sums
__global__ void kSpatialNorm(float* __restrict__ snorm) {
  int p = blockIdx.x * blockDim.x + threadIdx.x;
  int y = p / WW, x = p % WW;
  float w[TAPS];
  gauss_weights(SIGMA_SP, w);
  float sv = 0.f, sh = 0.f;
#pragma unroll
  for (int t = 0; t < TAPS; ++t) {
    int yy = y + t - RAD, xx = x + t - RAD;
    if (yy >= 0 && yy < HH) sv += w[t];
    if (xx >= 0 && xx < WW) sh += w[t];
  }
  snorm[p] = sv * sh;
}

// --------------------------- separable blur --------------------------------

// vertical pass; optional per-pixel guide multiply on load (bilateral input)
__global__ void kBlurV(const float* __restrict__ in, float* __restrict__ out,
                       const float* __restrict__ guide, float sigma) {
  size_t idx = (size_t)blockIdx.x * blockDim.x + threadIdx.x;
  int c = (int)(idx / NPIX);
  int p = (int)(idx % NPIX);
  int y = p / WW, x = p % WW;
  float w[TAPS];
  gauss_weights(sigma, w);
  // pull the bottom-of-stencil row toward the caches ahead of the tap loop
  if (y + RAD < HH) __builtin_prefetch(&in[(size_t)c * NPIX + (y + RAD) * WW + x], 0, 1);
  float acc = 0.f;
#pragma unroll
  for (int t = 0; t < TAPS; ++t) {
    int yy = y + t - RAD;
    if (yy >= 0 && yy < HH) {
      int pp = yy * WW + x;
      float v = in[(size_t)c * NPIX + pp];
      if (guide) v *= guide[pp];
      acc += w[t] * v;
    }
  }
  out[idx] = acc;
}

// horizontal pass, LDS tile filled with CDNA5 async global->LDS copies:
// 256-pixel segment + 7-pixel halos per block
__global__ void kBlurH(const float* __restrict__ in, float* __restrict__ out,
                       float sigma) {
  __shared__ float tile[256 + 2 * RAD];
  int b = blockIdx.x;
  int seg = b & 1;                 // 512 / 256 = 2 segments per row
  int rowIdx = b >> 1;             // global row across channels: c*HH + y
  int x0 = seg * 256;
  size_t base = (size_t)rowIdx * WW;
  int tid = threadIdx.x;

  // main body: one async dword per lane, no VGPR staging
  asyncLoadLdsB32(ldsAddr(&tile[RAD + tid]), in + base + x0 + tid);
  // halos (zero outside the row)
  if (tid < RAD) {
    int gl = x0 - RAD + tid;
    if (gl >= 0) asyncLoadLdsB32(ldsAddr(&tile[tid]), in + base + gl);
    else         tile[tid] = 0.f;
    int gr = x0 + 256 + tid;
    if (gr < WW) asyncLoadLdsB32(ldsAddr(&tile[RAD + 256 + tid]), in + base + gr);
    else         tile[RAD + 256 + tid] = 0.f;
  }
  waitAsync0();
  __syncthreads();

  float w[TAPS];
  gauss_weights(sigma, w);
  float acc = 0.f;
#pragma unroll
  for (int t = 0; t < TAPS; ++t) acc += w[t] * tile[tid + t];
  out[base + x0 + tid] = acc;
}

// --------------------------- per-iteration kernels -------------------------

__global__ void kSoftmax(const float* __restrict__ q, float* __restrict__ sm) {
  int p = blockIdx.x * blockDim.x + threadIdx.x;
  float m = -1e30f;
#pragma unroll
  for (int c = 0; c < CCH; ++c) m = fmaxf(m, q[(size_t)c * NPIX + p]);
  float s = 0.f;
  float e[CCH];
#pragma unroll
  for (int c = 0; c < CCH; ++c) {
    e[c] = __expf(q[(size_t)c * NPIX + p] - m);
    s += e[c];
  }
  float inv = 1.f / s;
#pragma unroll
  for (int c = 0; c < CCH; ++c) sm[(size_t)c * NPIX + p] = e[c] * inv;
}

// Normalize both filter outputs and pack pixel-major f16 with 32 padded
// channels -> B-fragment layout for WMMA (lane-contiguous 32B loads).
__global__ void kPack(const float* __restrict__ sblur, const float* __restrict__ bblur,
                      const float* __restrict__ snorm, const float* __restrict__ bnorm,
                      _Float16* __restrict__ Ps, _Float16* __restrict__ Pb) {
  size_t idx = (size_t)blockIdx.x * blockDim.x + threadIdx.x;   // over NPIX*32
  int p = (int)(idx >> 5);
  int cp = (int)(idx & 31);
  float vs = 0.f, vb = 0.f;
  if (cp < CCH) {
    size_t q = (size_t)cp * NPIX + p;
    vs = sblur[q] / snorm[p];
    vb = bblur[q] / bnorm[p];
  }
  Ps[idx] = (_Float16)vs;
  Pb[idx] = (_Float16)vb;
}

// A_np = qc + maxlab - l_eq (per pixel, all channels)
__global__ void kAnp(const float* __restrict__ q, const int* __restrict__ spm,
                     const int* __restrict__ spIdx, int iter,
                     float* __restrict__ Anp) {
  int p = blockIdx.x * blockDim.x + threadIdx.x;
  int y = p / WW, x = p % WW;
  bool cond = (spm[x * WW + y] == spIdx[iter]);   // transposed sp_map indexing
  float maxq = -1e30f;
#pragma unroll
  for (int c = 0; c < CCH; ++c) maxq = fmaxf(maxq, q[(size_t)c * NPIX + p]);
#pragma unroll
  for (int c = 0; c < CCH; ++c) {
    size_t idx = (size_t)c * NPIX + p;
    float qv = q[idx];
    float a = 0.f;
    if (cond) {
      float leq = (qv == maxq) ? qv : 0.f;
      a = qv + maxq - leq;
    }
    Anp[idx] = a;
  }
}

// stage-1 online logsumexp partials: z in {qc, A_np, cond2*q}
__global__ void kLse1(const float* __restrict__ q, const float* __restrict__ Anp,
                      const int* __restrict__ spm, const int* __restrict__ spIdx,
                      int iter, float* __restrict__ partM, float* __restrict__ partS) {
  __shared__ float sm_[256], ss_[256];
  int c = blockIdx.y;
  int z = blockIdx.z;
  int tid = threadIdx.x;
  int target = spIdx[iter] + (z == 2 ? 1 : 0);
  float m = -1e30f, s = 0.f;
  for (int p = blockIdx.x * 256 + tid; p < NPIX; p += 64 * 256) {
    float x;
    if (z == 1) {
      x = Anp[(size_t)c * NPIX + p];
    } else {
      int y = p / WW, xx = p % WW;
      bool cond = (spm[xx * WW + y] == target);
      x = cond ? q[(size_t)c * NPIX + p] : 0.f;
    }
    if (x > m) { s = s * __expf(m - x) + 1.f; m = x; }
    else       { s += __expf(x - m); }
  }
  sm_[tid] = m; ss_[tid] = s;
  __syncthreads();
  for (int off = 128; off > 0; off >>= 1) {
    if (tid < off) {
      float m2 = sm_[tid + off], s2 = ss_[tid + off];
      float M = fmaxf(sm_[tid], m2);
      ss_[tid] = ss_[tid] * __expf(sm_[tid] - M) + s2 * __expf(m2 - M);
      sm_[tid] = M;
    }
    __syncthreads();
  }
  if (tid == 0) {
    int slot = (z * CCH + c) * 64 + blockIdx.x;
    partM[slot] = sm_[0];
    partS[slot] = ss_[0];
  }
}

// stage-2 merge -> lse[z][c]
__global__ void kLse2(const float* __restrict__ partM, const float* __restrict__ partS,
                      float* __restrict__ lse) {
  __shared__ float sm_[64], ss_[64];
  int c = blockIdx.x, z = blockIdx.y;
  int tid = threadIdx.x;
  int slot = (z * CCH + c) * 64 + tid;
  sm_[tid] = partM[slot]; ss_[tid] = partS[slot];
  __syncthreads();
  for (int off = 32; off > 0; off >>= 1) {
    if (tid < off) {
      float m2 = sm_[tid + off], s2 = ss_[tid + off];
      float M = fmaxf(sm_[tid], m2);
      ss_[tid] = ss_[tid] * __expf(sm_[tid] - M) + s2 * __expf(m2 - M);
      sm_[tid] = M;
    }
    __syncthreads();
  }
  if (tid == 0) lse[z * CCH + c] = sm_[0] + __logf(ss_[0]);
}

// assemble combined relational update (cru)
__global__ void kCru(const float* __restrict__ q, const float* __restrict__ Anp,
                     const int* __restrict__ spm, const int* __restrict__ spIdx,
                     int iter, const float* __restrict__ lse,
                     const float* __restrict__ lowW, const float* __restrict__ highW,
                     float* __restrict__ cru) {
  int p = blockIdx.x * blockDim.x + threadIdx.x;
  int y = p / WW, x = p % WW;
  int base = spIdx[iter];
  bool cond  = (spm[x * WW + y] == base);
  bool cond2 = (spm[x * WW + y] == base + 1);
  float h0 = highW[0], h1 = highW[1], h2 = highW[2];
#pragma unroll
  for (int c = 0; c < CCH; ++c) {
    size_t idx = (size_t)c * NPIX + p;
    float qv = q[idx];
    float an = Anp[idx];
    float prod_sp  = cond  ? lse[0 * CCH + c] : 0.f;
    float prod_io  = cond  ? lse[1 * CCH + c] : 0.f;
    float prod_att = prod_sp + (cond2 ? lse[2 * CCH + c] : 0.f);
    float qmod = qv + ((qv == 0.f) ? 1.f : 0.f);
    float qst  = an + ((an == 0.f) ? 1.f : 0.f);
    float ft1 = prod_sp / qmod;
    float ft2 = prod_io / qst;
    float ft3 = prod_att / qmod;
    float t1 = lowW[0 * CCH + c] * ft1 + h0 * (1.f - ft1);
    float t2 = lowW[1 * CCH + c] * ft2 + h1 * (1.f - ft2);
    float t3 = lowW[2 * CCH + c] * ft3 + h2 * (1.f - ft3);
    cru[idx] = t1 + t2 + t3;
  }
}

// --------------------------- WMMA GEMM + update ----------------------------
// q = u - (M1 @ Ps + M2 @ Pb) - cru, 32x32 A (padded) x 32x16 pixel tiles.
__global__ void kGemmUpdate(const _Float16* __restrict__ Ps,
                            const _Float16* __restrict__ Pb,
                            const _Float16* __restrict__ M1,
                            const _Float16* __restrict__ M2,
                            const float* __restrict__ u,
                            const float* __restrict__ cru,
                            float* __restrict__ q) {
  const int lane = threadIdx.x & 31;
  const int hf   = lane >> 4;          // lane half (wave32)
  const int l16  = lane & 15;
  const int wave = blockIdx.x * (blockDim.x >> 5) + (threadIdx.x >> 5);
  const int nWaves = gridDim.x * (blockDim.x >> 5);

  // A-fragment k order chosen here (contiguous, same as B): lane%16 = row M, elem e -> K:
  //   K = e + 16*half
  v16h a1[2], a2[2];
#pragma unroll
  for (int mt = 0; mt < 2; ++mt) {
    int row = mt * 16 + l16;
#pragma unroll
    for (int e = 0; e < 16; ++e) {
      int kk = e + 16 * hf;
      a1[mt][e] = M1[row * 32 + kk];
      a2[mt][e] = M2[row * 32 + kk];
    }
  }

  const int nTiles = NPIX / 16;        // 16384, divisible by nWaves (4096)
  for (int t = wave; t < nTiles; t += nWaves) {
    int pix = t * 16 + l16;            // column N of this lane
    // B-fragment: lane = column, K 0-15 in lane half 0, K 16-31 in half 1
    const v16h bs = *(const v16h*)(Ps + (size_t)pix * 32 + 16 * hf);
    const v16h bb = *(const v16h*)(Pb + (size_t)pix * 32 + 16 * hf);
#pragma unroll
    for (int mt = 0; mt < 2; ++mt) {
      v8f acc = {};
      acc = __builtin_amdgcn_wmma_f32_16x16x32_f16(false, a1[mt], false, bs,
                                                   (short)0, acc, false, false);
      acc = __builtin_amdgcn_wmma_f32_16x16x32_f16(false, a2[mt], false, bb,
                                                   (short)0, acc, false, false);
#pragma unroll
      for (int v = 0; v < 8; ++v) {
        int row = mt * 16 + 8 * hf + v;        // D: VGPR v, lane half -> M
        if (row < CCH) {
          size_t idx = (size_t)row * NPIX + pix;
          q[idx] = u[idx] - acc[v] - cru[idx];
        }
      }
    }
  }
}

// planar CHW -> HWC output
__global__ void kOut(const float* __restrict__ q, float* __restrict__ out) {
  size_t idx = (size_t)blockIdx.x * blockDim.x + threadIdx.x;
  int c = (int)(idx / NPIX);
  int p = (int)(idx % NPIX);
  out[(size_t)p * CCH + c] = q[idx];
}

// ---------------------------------------------------------------------------

extern "C" void kernel_launch(void* const* d_in, const int* in_sizes, int n_in,
                              void* d_out, int out_size, void* d_ws, size_t ws_size,
                              hipStream_t stream) {
  const float* unaries = (const float*)d_in[0];
  const float* rgb     = (const float*)d_in[1];
  const int*   spm     = (const int*)d_in[2];
  const int*   spIdx   = (const int*)d_in[3];
  const float* Ws      = (const float*)d_in[4];
  const float* Wb      = (const float*)d_in[5];
  const float* lowW    = (const float*)d_in[6];
  const float* highW   = (const float*)d_in[7];
  const float* compat  = (const float*)d_in[8];
  float* out = (float*)d_out;

  float* W = (float*)d_ws;
  size_t off = 0;
  float* u     = W + off; off += CN;
  float* q     = W + off; off += CN;
  float* sm    = W + off; off += CN;
  float* tmp   = W + off; off += CN;
  float* sblur = W + off; off += CN;   // reused as A_np after packing
  float* bblur = W + off; off += CN;   // reused as cru after packing
  float* guide = W + off; off += NPIX;
  float* snorm = W + off; off += NPIX;
  float* bnorm = W + off; off += NPIX;
  float* tmpN  = W + off; off += NPIX;
  float* partM = W + off; off += 3 * CCH * 64;
  float* partS = W + off; off += 3 * CCH * 64;
  float* lse   = W + off; off += 64;   // 3*21 used, padded for alignment
  _Float16* M1h = (_Float16*)(W + off); off += 1024;  // 2048 halves
  _Float16* M2h = (_Float16*)(W + off); off += 1024;
  _Float16* Ps  = (_Float16*)(W + off); off += (size_t)NPIX * 16;  // NPIX*32 halves
  _Float16* Pb  = (_Float16*)(W + off); off += (size_t)NPIX * 16;

  const int B = 256;
  const int gN  = NPIX / B;        // 1024
  const int gCN = CN / B;          // 21504
  const int gP  = (NPIX * 32) / B; // 32768

  // one-time prep
  kPrepGuide<<<gN, B, 0, stream>>>(rgb, guide);
  kPrepU<<<gCN, B, 0, stream>>>(unaries, u, q);
  kMatPrep<<<8, B, 0, stream>>>(compat, Ws, Wb, M1h, M2h);
  kSpatialNorm<<<gN, B, 0, stream>>>(snorm);
  // bilateral norm = full blur of guide (single channel)
  kBlurV<<<gN, B, 0, stream>>>(guide, tmpN, nullptr, SIGMA_BI);
  kBlurH<<<1 * HH * 2, B, 0, stream>>>(tmpN, bnorm, SIGMA_BI);

  for (int it = 0; it < NUM_ITER; ++it) {
    kSoftmax<<<gN, B, 0, stream>>>(q, sm);
    // spatial filter
    kBlurV<<<gCN, B, 0, stream>>>(sm, tmp, nullptr, SIGMA_SP);
    kBlurH<<<CCH * HH * 2, B, 0, stream>>>(tmp, sblur, SIGMA_SP);
    // bilateral filter (guide multiplied on load)
    kBlurV<<<gCN, B, 0, stream>>>(sm, tmp, guide, SIGMA_BI);
    kBlurH<<<CCH * HH * 2, B, 0, stream>>>(tmp, bblur, SIGMA_BI);
    // normalize + pack f16 pixel-major for WMMA B-fragments
    kPack<<<gP, B, 0, stream>>>(sblur, bblur, snorm, bnorm, Ps, Pb);
    // combined relational update (sblur/bblur now free -> A_np / cru)
    kAnp<<<gN, B, 0, stream>>>(q, spm, spIdx, it, sblur);
    kLse1<<<dim3(64, CCH, 3), B, 0, stream>>>(q, sblur, spm, spIdx, it, partM, partS);
    kLse2<<<dim3(CCH, 3), 64, 0, stream>>>(partM, partS, lse);
    kCru<<<gN, B, 0, stream>>>(q, sblur, spm, spIdx, it, lse, lowW, highW, bblur);
    // fused WMMA matmul + q update: q = u - compat@(Ws@s + Wb@b) - cru
    kGemmUpdate<<<1024, 128, 0, stream>>>(Ps, Pb, M1h, M2h, u, bblur, q);
  }

  kOut<<<gCN, B, 0, stream>>>(q, out);
}
